// ImprovedGATModel_987842478162
// MI455X (gfx1250) — compile-verified, matrix operand order checked by reading, not run on hardware
//
#include <hip/hip_runtime.h>
#include <hip/hip_bf16.h>
#include <stdint.h>

// ---------------------------------------------------------------------------
// GAT (3-layer, PyG-style) for MI455X / gfx1250.
//  * GEMMs via v_wmma_f32_16x16x32_bf16 (wave32, one 16x16 tile per wave).
//  * Edge softmax via 3 atomic passes (L2-resident: 51.2MB features << 192MB L2).
//  * BN bias cancels analytically (b1/b2 unused); b3 applied at the end.
// ---------------------------------------------------------------------------

typedef __bf16 bf16_t;
typedef __attribute__((ext_vector_type(8)))  bf16_t v8bf;
typedef __attribute__((ext_vector_type(16))) bf16_t v16bf;
typedef __attribute__((ext_vector_type(8)))  float  v8f;

#define NS_ATT 0.2f
#define NS_ACT 0.01f
#define BN_EPS 1e-5f

static __device__ __forceinline__ float lrelu(float x, float s) {
  return x >= 0.f ? x : s * x;
}
// Monotonic unsigned key for float atomicMax (total order, works for +/-).
static __device__ __forceinline__ unsigned fkey(float f) {
  unsigned u = __float_as_uint(f);
  return (u & 0x80000000u) ? ~u : (u | 0x80000000u);
}
static __device__ __forceinline__ float keyf(unsigned k) {
  return (k & 0x80000000u) ? __uint_as_float(k & 0x7fffffffu)
                           : __uint_as_float(~k);
}

// ---- conversions ----------------------------------------------------------
__global__ void k_cvt_bf16(const float* __restrict__ in, bf16_t* __restrict__ out, int n) {
  int i = blockIdx.x * blockDim.x + threadIdx.x;
  if (i < n) out[i] = (bf16_t)in[i];
}

// W [K x N] f32 -> WT [N x K] bf16 (so WMMA B columns are contiguous in K)
__global__ void k_transpose_bf16(const float* __restrict__ W, bf16_t* __restrict__ WT,
                                 int K, int N) {
  int i = blockIdx.x * blockDim.x + threadIdx.x;
  if (i >= K * N) return;
  int n = i / K, k = i - n * K;
  WT[i] = (bf16_t)W[k * N + n];
}

// ---- WMMA GEMM: C[MxN] f32 = A[MxK] bf16 * B, with B given as BT[N x K] ----
// One wave32 computes one 16x16 tile; K consumed 32 at a time.
// A lane layout (16-bit A 16x32): lane L row = L&15, kbase = (L>>4)*8,
//   elems 0..7 -> K=kbase..kbase+7, elems 8..15 -> K=kbase+16..kbase+23.
// B mirrors with N in place of M (lane holds a column of B).
__global__ void k_gemm_bf16_wmma(const bf16_t* __restrict__ A,
                                 const bf16_t* __restrict__ BT,
                                 float* __restrict__ C, int M, int N, int K) {
  int wave = (int)((blockIdx.x * (long long)blockDim.x + threadIdx.x) >> 5);
  int lane = threadIdx.x & 31;
  int ntiles = N >> 4;
  int tm = wave / ntiles;
  int tn = wave - tm * ntiles;
  if (tm >= (M >> 4)) return;  // uniform per wave: EXEC stays all-1s for WMMA

  int r  = (tm << 4) + (lane & 15);   // A row this lane feeds
  int cn = (tn << 4) + (lane & 15);   // B column this lane feeds
  int kb = (lane >> 4) << 3;          // 0 or 8
  const bf16_t* ap = A  + (size_t)r  * K + kb;
  const bf16_t* bp = BT + (size_t)cn * K + kb;

  v8f acc = {};
  for (int k0 = 0; k0 < K; k0 += 32) {
    v8bf a_lo = *(const v8bf*)(ap + k0);
    v8bf a_hi = *(const v8bf*)(ap + k0 + 16);
    v8bf b_lo = *(const v8bf*)(bp + k0);
    v8bf b_hi = *(const v8bf*)(bp + k0 + 16);
    v16bf a = __builtin_shufflevector(a_lo, a_hi, 0,1,2,3,4,5,6,7,8,9,10,11,12,13,14,15);
    v16bf b = __builtin_shufflevector(b_lo, b_hi, 0,1,2,3,4,5,6,7,8,9,10,11,12,13,14,15);
    acc = __builtin_amdgcn_wmma_f32_16x16x32_bf16(false, a, false, b,
                                                  (short)0, acc, false, false);
  }
  // C/D layout: VGPR r at lane L -> M = r + 8*(L>>4), N = L&15.
  int mrow = (tm << 4) + ((lane >> 4) << 3);
  float* cp = C + (size_t)mrow * N + (tn << 4) + (lane & 15);
#pragma unroll
  for (int i = 0; i < 8; ++i) cp[(size_t)i * N] = acc[i];
}

// ---- per-node attention logits: als/ald [N*heads] -------------------------
__global__ void k_node_logits(const float* __restrict__ h, const float* __restrict__ a_s,
                              const float* __restrict__ a_d, float* __restrict__ als,
                              float* __restrict__ ald, int nodes, int heads, int ch) {
  int idx = blockIdx.x * blockDim.x + threadIdx.x;  // n*heads + hh
  if (idx >= nodes * heads) return;
  int hh = idx % heads;
  const float* hp = h + (size_t)idx * ch;  // h[n][hh][c] == h[(n*heads+hh)*ch + c]
  float s = 0.f, d = 0.f;
  for (int c = 0; c < ch; ++c) {
    float v = hp[c];
    s += v * a_s[hh * ch + c];
    d += v * a_d[hh * ch + c];
  }
  als[idx] = s;
  ald[idx] = d;
}

// ---- edge pass 1: segment max over dst (atomicMax on ordered key) ---------
__global__ void k_edge_max(const int* __restrict__ ei, const float* __restrict__ als,
                           const float* __restrict__ ald, unsigned* __restrict__ mkey,
                           int E, int nodes, int heads) {
  long long idx = (long long)blockIdx.x * blockDim.x + threadIdx.x;
  long long tot = (long long)(E + nodes) * heads;
  if (idx >= tot) return;
  int e = (int)(idx / heads), hh = (int)(idx % heads);
  int s = (e < E) ? ei[e] : (e - E);       // self-loops appended after edges
  int d = (e < E) ? ei[E + e] : (e - E);
  float l = lrelu(als[s * heads + hh] + ald[d * heads + hh], NS_ATT);
  atomicMax(&mkey[d * heads + hh], fkey(l));
}

// ---- edge pass 2: denom[dst] += exp(l - m[dst]) ---------------------------
__global__ void k_edge_denom(const int* __restrict__ ei, const float* __restrict__ als,
                             const float* __restrict__ ald, const unsigned* __restrict__ mkey,
                             float* __restrict__ denom, int E, int nodes, int heads) {
  long long idx = (long long)blockIdx.x * blockDim.x + threadIdx.x;
  long long tot = (long long)(E + nodes) * heads;
  if (idx >= tot) return;
  int e = (int)(idx / heads), hh = (int)(idx % heads);
  int s = (e < E) ? ei[e] : (e - E);
  int d = (e < E) ? ei[E + e] : (e - E);
  int nh = d * heads + hh;
  float l = lrelu(als[s * heads + hh] + ald[nh], NS_ATT);
  atomicAdd(&denom[nh], __expf(l - keyf(mkey[nh])));
}

// ---- edge pass 3: agg[dst] += alpha * h[src] (per channel) ----------------
__global__ void k_edge_aggregate(const int* __restrict__ ei, const float* __restrict__ als,
                                 const float* __restrict__ ald,
                                 const unsigned* __restrict__ mkey,
                                 const float* __restrict__ denom,
                                 const float* __restrict__ h, float* __restrict__ agg,
                                 int E, int nodes, int heads, int ch) {
  long long idx = (long long)blockIdx.x * blockDim.x + threadIdx.x;
  int HC = heads * ch;
  long long tot = (long long)(E + nodes) * HC;
  if (idx >= tot) return;
  int e = (int)(idx / HC);
  int j = (int)(idx - (long long)e * HC);
  int hh = j / ch;
  int s = (e < E) ? ei[e] : (e - E);
  int d = (e < E) ? ei[E + e] : (e - E);
  int nh = d * heads + hh;
  float l = lrelu(als[s * heads + hh] + ald[nh], NS_ATT);
  float alpha = __expf(l - keyf(mkey[nh])) / (denom[nh] + 1e-16f);
  atomicAdd(&agg[(size_t)d * HC + j], alpha * h[(size_t)s * HC + j]);
}

// ---- BatchNorm column stats -----------------------------------------------
__global__ void k_bn_stats(const float* __restrict__ x, float* __restrict__ mu,
                           float* __restrict__ rsig, int rows, int cols) {
  int col = blockIdx.x;
  int tid = threadIdx.x;
  float s = 0.f, s2 = 0.f;
  for (int r = tid; r < rows; r += blockDim.x) {
    float v = x[(size_t)r * cols + col];
    s += v;
    s2 += v * v;
  }
  __shared__ float sh[256], sh2[256];
  sh[tid] = s; sh2[tid] = s2;
  __syncthreads();
  for (int off = blockDim.x >> 1; off > 0; off >>= 1) {
    if (tid < off) { sh[tid] += sh[tid + off]; sh2[tid] += sh2[tid + off]; }
    __syncthreads();
  }
  if (tid == 0) {
    float m = sh[0] / rows;
    float var = sh2[0] / rows - m * m;
    mu[col] = m;
    rsig[col] = rsqrtf(var + BN_EPS);
  }
}

// ---- fused BN + LeakyReLU (+ optional residual add) -----------------------
__global__ void k_bn_act(const float* __restrict__ x, const float* __restrict__ mu,
                         const float* __restrict__ rsig, const float* __restrict__ gamma,
                         const float* __restrict__ beta, const float* __restrict__ residual,
                         float* __restrict__ out, int total, int cols) {
  int i = blockIdx.x * blockDim.x + threadIdx.x;
  if (i >= total) return;
  int c = i % cols;
  float v = lrelu((x[i] - mu[c]) * rsig[c] * gamma[c] + beta[c], NS_ACT);
  if (residual) v += residual[i];
  out[i] = v;
}

// ---- tiny f32 GEMM for layer 3 (256 -> 4) ---------------------------------
__global__ void k_gemm_small(const float* __restrict__ x, const float* __restrict__ W,
                             float* __restrict__ y, int rows, int K, int O) {
  int i = blockIdx.x * blockDim.x + threadIdx.x;
  if (i >= rows * O) return;
  int n = i / O, o = i - n * O;
  const float* xp = x + (size_t)n * K;
  float s = 0.f;
  for (int k = 0; k < K; ++k) s += xp[k] * W[k * O + o];
  y[i] = s;
}

__global__ void k_out_bias(const float* __restrict__ agg, const float* __restrict__ b,
                           float* __restrict__ out, int total, int O) {
  int i = blockIdx.x * blockDim.x + threadIdx.x;
  if (i >= total) return;
  out[i] = agg[i] + b[i % O];
}

// ---------------------------------------------------------------------------
extern "C" void kernel_launch(void* const* d_in, const int* in_sizes, int n_in,
                              void* d_out, int out_size, void* d_ws, size_t ws_size,
                              hipStream_t stream) {
  const float* x   = (const float*)d_in[0];
  const int*   ei  = (const int*)d_in[1];
  const float* W1  = (const float*)d_in[2];
  const float* as1 = (const float*)d_in[3];
  const float* ad1 = (const float*)d_in[4];
  const float* g1  = (const float*)d_in[6];
  const float* be1 = (const float*)d_in[7];
  const float* W2  = (const float*)d_in[8];
  const float* as2 = (const float*)d_in[9];
  const float* ad2 = (const float*)d_in[10];
  const float* g2  = (const float*)d_in[12];
  const float* be2 = (const float*)d_in[13];
  const float* W3  = (const float*)d_in[14];
  const float* as3 = (const float*)d_in[15];
  const float* ad3 = (const float*)d_in[16];
  const float* b3  = (const float*)d_in[17];
  (void)n_in; (void)out_size; (void)ws_size;

  const int IN = 128, HEADS = 8, CH = 32, HC = HEADS * CH, OUT = 4;
  const int N = in_sizes[0] / IN;  // 50000 (multiple of 16)
  const int E = in_sizes[1] / 2;   // 800000

  // ---- carve workspace (~190 MB) ----
  char* ws = (char*)d_ws;
  size_t off = 0;
  auto take = [&](size_t bytes) -> char* {
    char* p = ws + off;
    off += (bytes + 255) & ~(size_t)255;
    return p;
  };
  bf16_t*   bfbuf  = (bf16_t*)take((size_t)N * HC * 2);   // bf16 activations (reused)
  bf16_t*   w1t    = (bf16_t*)take((size_t)HC * IN * 2);
  bf16_t*   w2t    = (bf16_t*)take((size_t)HC * HC * 2);
  float*    buf1   = (float*)take((size_t)N * HC * 4);    // linear h / act2
  float*    buf2   = (float*)take((size_t)N * HC * 4);    // aggregated
  float*    buf3   = (float*)take((size_t)N * HC * 4);    // act1 (residual)
  float*    als    = (float*)take((size_t)N * HEADS * 4);
  float*    ald    = (float*)take((size_t)N * HEADS * 4);
  unsigned* mkey   = (unsigned*)take((size_t)N * HEADS * 4);
  float*    denom  = (float*)take((size_t)N * HEADS * 4);
  float*    mu     = (float*)take((size_t)HC * 4);
  float*    rsig   = (float*)take((size_t)HC * 4);
  float*    h3     = (float*)take((size_t)N * OUT * 4);
  float*    als3   = (float*)take((size_t)N * 4);
  float*    ald3   = (float*)take((size_t)N * 4);
  unsigned* mkey3  = (unsigned*)take((size_t)N * 4);
  float*    denom3 = (float*)take((size_t)N * 4);
  float*    agg3   = (float*)take((size_t)N * OUT * 4);

  auto blocks = [](long long n) { return dim3((unsigned)((n + 255) / 256)); };
  const dim3 T(256);

  // ---- prep: bf16 convert + weight transposes ----
  k_cvt_bf16<<<blocks((long long)N * IN), T, 0, stream>>>(x, bfbuf, N * IN);
  k_transpose_bf16<<<blocks(IN * HC), T, 0, stream>>>(W1, w1t, IN, HC);
  k_transpose_bf16<<<blocks(HC * HC), T, 0, stream>>>(W2, w2t, HC, HC);

  const long long gemm_thr = (long long)(N / 16) * (HC / 16) * 32;

  // ================= Layer 1: GATConv(128 -> 8x32) + BN + LReLU ============
  k_gemm_bf16_wmma<<<blocks(gemm_thr), T, 0, stream>>>(bfbuf, w1t, buf1, N, HC, IN);
  k_node_logits<<<blocks((long long)N * HEADS), T, 0, stream>>>(buf1, as1, ad1, als, ald, N, HEADS, CH);
  hipMemsetAsync(mkey, 0, (size_t)N * HEADS * 4, stream);
  hipMemsetAsync(denom, 0, (size_t)N * HEADS * 4, stream);
  hipMemsetAsync(buf2, 0, (size_t)N * HC * 4, stream);
  k_edge_max<<<blocks((long long)(E + N) * HEADS), T, 0, stream>>>(ei, als, ald, mkey, E, N, HEADS);
  k_edge_denom<<<blocks((long long)(E + N) * HEADS), T, 0, stream>>>(ei, als, ald, mkey, denom, E, N, HEADS);
  k_edge_aggregate<<<blocks((long long)(E + N) * HC), T, 0, stream>>>(ei, als, ald, mkey, denom,
                                                                      buf1, buf2, E, N, HEADS, CH);
  // bias b1 cancels inside BN: (x+b) - mean(x+b) == x - mean(x)
  k_bn_stats<<<dim3(HC), T, 0, stream>>>(buf2, mu, rsig, N, HC);
  k_bn_act<<<blocks((long long)N * HC), T, 0, stream>>>(buf2, mu, rsig, g1, be1, nullptr,
                                                        buf3, N * HC, HC);

  // ============ Layer 2: GATConv(256 -> 8x32) + BN + LReLU + residual ======
  k_cvt_bf16<<<blocks((long long)N * HC), T, 0, stream>>>(buf3, bfbuf, N * HC);
  k_gemm_bf16_wmma<<<blocks(gemm_thr), T, 0, stream>>>(bfbuf, w2t, buf1, N, HC, HC);
  k_node_logits<<<blocks((long long)N * HEADS), T, 0, stream>>>(buf1, as2, ad2, als, ald, N, HEADS, CH);
  hipMemsetAsync(mkey, 0, (size_t)N * HEADS * 4, stream);
  hipMemsetAsync(denom, 0, (size_t)N * HEADS * 4, stream);
  k_edge_max<<<blocks((long long)(E + N) * HEADS), T, 0, stream>>>(ei, als, ald, mkey, E, N, HEADS);
  k_edge_denom<<<blocks((long long)(E + N) * HEADS), T, 0, stream>>>(ei, als, ald, mkey, denom, E, N, HEADS);
  hipMemsetAsync(buf2, 0, (size_t)N * HC * 4, stream);
  k_edge_aggregate<<<blocks((long long)(E + N) * HC), T, 0, stream>>>(ei, als, ald, mkey, denom,
                                                                      buf1, buf2, E, N, HEADS, CH);
  k_bn_stats<<<dim3(HC), T, 0, stream>>>(buf2, mu, rsig, N, HC);
  // act2 = lrelu(bn(agg2)) + residual(act1); h2-lin in buf1 no longer needed
  k_bn_act<<<blocks((long long)N * HC), T, 0, stream>>>(buf2, mu, rsig, g2, be2, buf3,
                                                        buf1, N * HC, HC);

  // ================= Layer 3: GATConv(256 -> 4, heads=1) ===================
  k_gemm_small<<<blocks((long long)N * OUT), T, 0, stream>>>(buf1, W3, h3, N, HC, OUT);
  k_node_logits<<<blocks((long long)N), T, 0, stream>>>(h3, as3, ad3, als3, ald3, N, 1, OUT);
  hipMemsetAsync(mkey3, 0, (size_t)N * 4, stream);
  hipMemsetAsync(denom3, 0, (size_t)N * 4, stream);
  hipMemsetAsync(agg3, 0, (size_t)N * OUT * 4, stream);
  k_edge_max<<<blocks((long long)(E + N)), T, 0, stream>>>(ei, als3, ald3, mkey3, E, N, 1);
  k_edge_denom<<<blocks((long long)(E + N)), T, 0, stream>>>(ei, als3, ald3, mkey3, denom3, E, N, 1);
  k_edge_aggregate<<<blocks((long long)(E + N) * OUT), T, 0, stream>>>(ei, als3, ald3, mkey3, denom3,
                                                                       h3, agg3, E, N, 1, OUT);
  k_out_bias<<<blocks((long long)N * OUT), T, 0, stream>>>(agg3, b3, (float*)d_out, N * OUT, OUT);
}
